// PolynormerGraph_20349555048607
// MI455X (gfx1250) — matrix operand order repaired and checked by reading, not compiled, not run on hardware
//
#include <hip/hip_runtime.h>
#include <hip/hip_bf16.h>
#include <math.h>

// Problem constants (match reference)
#define DM   128
#define HH   8
#define CC   16
#define NN   50000
#define EE   800000
#define GG   512
#define OUTD 10

typedef __attribute__((ext_vector_type(16))) __bf16 v16bf;
typedef __attribute__((ext_vector_type(8)))  float  v8f;
typedef __attribute__((ext_vector_type(4)))  unsigned int u32x4;
typedef __attribute__((ext_vector_type(8)))  int  i32x8;
typedef __attribute__((ext_vector_type(4)))  int  i32x4;

__device__ __forceinline__ float waveRed(float v) {
#pragma unroll
  for (int off = 16; off >= 1; off >>= 1) v += __shfl_xor(v, off, 32);
  return v;
}

__device__ __forceinline__ void atomicMaxF(float* addr, float val) {
  if (val >= 0.f) atomicMax((int*)addr, __float_as_int(val));
  else            atomicMin((unsigned int*)addr, __float_as_uint(val));
}

#define HAVE_TDM (__has_builtin(__builtin_amdgcn_tensor_load_to_lds) && \
                  __has_builtin(__builtin_amdgcn_s_wait_tensorcnt))

// ---------------------------------------------------------------------------
// out[nrows,128] = ACT( A[nrows,128] @ W[128,128] + bias )
// ACT: 0=none, 1=relu, 2=sigmoid.  Requires nrows % 16 == 0.
// W staged into LDS via the Tensor Data Mover (TDM), then converted to a
// transposed bf16 layout feeding v_wmma_f32_16x16x32_bf16.
// One wave -> 16 output rows; block = 4 waves = 64 rows.
// ---------------------------------------------------------------------------
template<int ACT>
__global__ void __launch_bounds__(128)
gemm128_wmma(const float* __restrict__ A, const float* __restrict__ W,
             const float* __restrict__ bias, float* __restrict__ out, int nrows)
{
  __shared__ __bf16 ldsB[DM * DM];           // ldsB[n*128 + k] = bf16(W[k][n])
#if HAVE_TDM
  __shared__ float ldsW[DM * DM];            // raw fp32 W staged by TDM (64 KB)
#endif

  const int t    = threadIdx.x;
  const int wave = t >> 5;
  const int lane = t & 31;

#if HAVE_TDM
  {
    // Tensor DMA Descriptor (D#): 2-D tensor == tile, 128x128 x 4B, stride 128.
    // Group0: count=1 | lds_addr | global_addr[56:0] | type=2 (ISA 8.3)
    // Group1: data_size=4B, tensor_dim0/1=128, tile_dim0/1=128, stride0=128 (ISA 8.4)
    unsigned long long ga = (unsigned long long)(uintptr_t)W;
    unsigned int ldsOff = (unsigned int)(uintptr_t)&ldsW[0]; // LDS aperture low bits = byte offset
    u32x4 g0;
    g0[0] = 1u;                                                   // count=1, user D#
    g0[1] = ldsOff;                                               // lds_addr
    g0[2] = (unsigned int)ga;                                     // global_addr[31:0]
    g0[3] = (unsigned int)((ga >> 32) & 0x01FFFFFFull) | (2u << 30); // addr[56:32] | type=2
    i32x8 g1;
    g1[0] = (int)(2u << 16);          // workgroup_mask=0, data_size=2 (4 bytes)
    g1[1] = (int)(128u << 16);        // tensor_dim0[15:0] in bits 63:48
    g1[2] = (int)(128u << 16);        // tensor_dim1[15:0] in bits 95:80
    g1[3] = (int)(128u << 16);        // tile_dim0 in bits 127:112
    g1[4] = 128;                      // tile_dim1 in bits 143:128 (tile_dim2=0)
    g1[5] = 128;                      // tensor_dim0_stride[31:0]
    g1[6] = 0;                        // stride hi / dim1_stride (2-D: unused)
    g1[7] = 0;
    i32x4 z4 = {0, 0, 0, 0};          // groups 2/3 disabled (<=2-D tensor)
    i32x8 z8 = {0, 0, 0, 0, 0, 0, 0, 0};
    __builtin_amdgcn_tensor_load_to_lds(g0, g1, z4, z4, z8, 0);
    __builtin_amdgcn_s_wait_tensorcnt((short)0);  // each wave's own copy complete
  }
  for (int i = t; i < DM * DM; i += 128) {
    int k = i >> 7, n = i & (DM - 1);
    ldsB[n * DM + k] = (__bf16)ldsW[i];      // native v_cvt bf16
  }
#else
  for (int i = t; i < DM * DM; i += 128) {
    int k = i >> 7, n = i & (DM - 1);
    ldsB[n * DM + k] = (__bf16)W[i];
  }
#endif
  __syncthreads();

  const int row0 = blockIdx.x * 64 + wave * 16;
  if (row0 >= nrows) return;                 // wave-uniform; tiles are always full

  // A fragments: 16-bit A 16x32 layout. lanes 0-15: K in {0-7,16-23};
  // lanes 16-31: K in {8-15,24-31}; packed pair (k0,k0+1) per dword.
  const int m = lane & 15;
  const float* arow = A + (long)(row0 + m) * DM;
  v16bf afrag[4];
#pragma unroll
  for (int kc = 0; kc < 4; ++kc) {
    v16bf p;
#pragma unroll
    for (int j = 0; j < 8; ++j) {
      int k0 = kc * 32 + 2 * j + ((j >= 4) ? 8 : 0) + ((lane >= 16) ? 8 : 0);
      float2 f = *(const float2*)(arow + k0);
      p[2 * j]     = (__bf16)f.x;            // pairs -> v_cvt_pk_bf16_f32
      p[2 * j + 1] = (__bf16)f.y;
    }
    afrag[kc] = p;
  }

  const int nIdx = lane & 15;
  const int kOff = (lane >= 16) ? 8 : 0;     // B 32x16: lane halves split K
  float bcol[8];
#pragma unroll
  for (int ct = 0; ct < 8; ++ct) bcol[ct] = bias ? bias[ct * 16 + nIdx] : 0.f;

  const int rbase = row0 + ((lane >= 16) ? 8 : 0);  // C/D layout: M = r + 8*(lane>=16)
  for (int ct = 0; ct < 8; ++ct) {
    const int col = ct * 16 + nIdx;
    v8f acc = {};
#pragma unroll
    for (int kc = 0; kc < 4; ++kc) {
      v16bf pb;
#pragma unroll
      for (int v = 0; v < 8; ++v) {
        int k = kc * 32 + 2 * v + ((v >= 4) ? 8 : 0) + kOff;      // k even -> 4B aligned pair load
        *((unsigned int*)&pb + v) = *(const unsigned int*)&ldsB[col * DM + k];
      }
      acc = __builtin_amdgcn_wmma_f32_16x16x32_bf16(
              false, afrag[kc], false, pb, (short)0, acc, false, false);
    }
#pragma unroll
    for (int r = 0; r < 8; ++r) {
      float v = acc[r] + bcol[ct];
      if (ACT == 1)      v = fmaxf(v, 0.f);
      else if (ACT == 2) v = 1.f / (1.f + __expf(-v));
      out[(long)(rbase + r) * DM + col] = v;
    }
  }
}

// ---------------------------------------------------------------------------
__global__ void fill_k(float* __restrict__ p, long long n, float v) {
  long long i = (long long)blockIdx.x * blockDim.x + threadIdx.x;
  if (i < n) p[i] = v;
}

// es/ed: per (node, head) attention logits
__global__ void edge_scores_k(const float* __restrict__ xw,
                              const float* __restrict__ a_s, const float* __restrict__ a_d,
                              float* __restrict__ es, float* __restrict__ ed, int n)
{
  int i = blockIdx.x * blockDim.x + threadIdx.x;
  if (i >= n * HH) return;
  int node = i / HH, h = i % HH;
  const float* xr = xw + (long)node * DM + h * CC;
  float s = 0.f, d2 = 0.f;
#pragma unroll
  for (int c = 0; c < CC; ++c) { float v = xr[c]; s += v * a_s[h*CC+c]; d2 += v * a_d[h*CC+c]; }
  es[i] = s; ed[i] = d2;
}

__global__ void edge_pass1(const int* __restrict__ ei, const float* __restrict__ es,
                           const float* __restrict__ ed, float* __restrict__ ebuf,
                           float* __restrict__ emax, int nE, int nN)
{
  long long i = (long long)blockIdx.x * blockDim.x + threadIdx.x;
  long long tot = (long long)(nE + nN) * HH;
  if (i >= tot) return;
  long long eidx = i / HH; int h = (int)(i % HH);
  int s, d;
  if (eidx < nE) { s = ei[eidx]; d = ei[(long long)nE + eidx]; }
  else           { s = d = (int)(eidx - nE); }
  float e = es[(long)s * HH + h] + ed[(long)d * HH + h];
  e = (e > 0.f) ? e : 0.2f * e;              // leaky_relu(0.2)
  ebuf[i] = e;
  atomicMaxF(&emax[(long)d * HH + h], e);
}

__global__ void edge_pass2(const int* __restrict__ ei, float* __restrict__ ebuf,
                           const float* __restrict__ emax, float* __restrict__ den,
                           int nE, int nN)
{
  long long i = (long long)blockIdx.x * blockDim.x + threadIdx.x;
  long long tot = (long long)(nE + nN) * HH;
  if (i >= tot) return;
  long long eidx = i / HH; int h = (int)(i % HH);
  int d;
  if (eidx < nE) d = ei[(long long)nE + eidx];
  else           d = (int)(eidx - nE);
  float ee = __expf(ebuf[i] - emax[(long)d * HH + h]);
  ebuf[i] = ee;
  atomicAdd(&den[(long)d * HH + h], ee);
}

// acc (pre-seeded with x@Wl+bl) += alpha * xw[src]
__global__ void edge_pass3(const int* __restrict__ ei, const float* __restrict__ ebuf,
                           const float* __restrict__ den, const float* __restrict__ xw,
                           float* __restrict__ acc, int nE, int nN)
{
  long long i = (long long)blockIdx.x * blockDim.x + threadIdx.x;
  long long tot = (long long)(nE + nN) * DM;
  if (i >= tot) return;
  long long eidx = i >> 7; int dcol = (int)(i & (DM - 1)); int h = dcol >> 4;
  int s, d;
  if (eidx < nE) { s = ei[eidx]; d = ei[(long long)nE + eidx]; }
  else           { s = d = (int)(eidx - nE); }
  float alpha = ebuf[eidx * HH + h] / den[(long)d * HH + h];
  atomicAdd(&acc[(long)d * DM + dcol], alpha * xw[(long)s * DM + dcol]);
}

// x = (1-sigmoid(beta))*LN(h * relu(gat+b_gat)) + sigmoid(beta)*relu(...); xlocal += x
__global__ void __launch_bounds__(256)
local_fuse(const float* __restrict__ gat, const float* __restrict__ b_gat,
           const float* __restrict__ hbuf, const float* __restrict__ betas,
           const float* __restrict__ lng, const float* __restrict__ lnb,
           float* __restrict__ x, float* __restrict__ xlocal, int n)
{
  int wave = threadIdx.x >> 5, lane = threadIdx.x & 31;
  int row = blockIdx.x * 8 + wave;
  if (row >= n) return;
  long base = (long)row * DM;
  float tval[4], y[4]; float s = 0.f;
#pragma unroll
  for (int j = 0; j < 4; ++j) {
    int d = lane * 4 + j;
    float tv = fmaxf(gat[base + d] + b_gat[d], 0.f);
    tval[j] = tv;
    y[j] = hbuf[base + d] * tv;
    s += y[j];
  }
  float mean = waveRed(s) * (1.f / DM);
  float vs = 0.f;
#pragma unroll
  for (int j = 0; j < 4; ++j) { float dl = y[j] - mean; vs += dl * dl; }
  float rs = rsqrtf(waveRed(vs) * (1.f / DM) + 1e-5f);
#pragma unroll
  for (int j = 0; j < 4; ++j) {
    int d = lane * 4 + j;
    float ln = (y[j] - mean) * rs * lng[d] + lnb[d];
    float beta = 1.f / (1.f + __expf(-betas[d]));
    float xv = (1.f - beta) * ln + beta * tval[j];
    x[base + d] = xv;
    xlocal[base + d] += xv;
  }
}

__global__ void pool_k(const float* __restrict__ xl, const int* __restrict__ batch,
                       float* __restrict__ xp, float* __restrict__ cnt, int n)
{
  long long i = (long long)blockIdx.x * blockDim.x + threadIdx.x;
  if (i >= (long long)n * DM) return;
  int node = (int)(i >> 7), d = (int)(i & (DM - 1));
  int g = batch[node];
  atomicAdd(&xp[(long)g * DM + d], xl[i]);
  if (d == 0) atomicAdd(&cnt[g], 1.f);
}

__global__ void pool_ln_k(const float* __restrict__ xp, const float* __restrict__ cnt,
                          const float* __restrict__ g_, const float* __restrict__ b_,
                          float* __restrict__ out, int nG)
{
  int wave = threadIdx.x >> 5, lane = threadIdx.x & 31;
  int g = blockIdx.x * 8 + wave;
  if (g >= nG) return;
  long base = (long)g * DM;
  float inv = 1.f / fmaxf(cnt[g], 1.f);
  float y[4]; float s = 0.f;
#pragma unroll
  for (int j = 0; j < 4; ++j) { y[j] = xp[base + lane*4 + j] * inv; s += y[j]; }
  float mean = waveRed(s) * (1.f / DM);
  float vs = 0.f;
#pragma unroll
  for (int j = 0; j < 4; ++j) { float dl = y[j] - mean; vs += dl * dl; }
  float rs = rsqrtf(waveRed(vs) * (1.f / DM) + 1e-5f);
#pragma unroll
  for (int j = 0; j < 4; ++j) {
    int d = lane * 4 + j;
    out[base + d] = (y[j] - mean) * rs * g_[d] + b_[d];
  }
}

// kv[dd,m,hh] = sum_g k[g,dd,hh] * v[g,m,hh]   (k[g,c,h] = gk[g*128 + c*8 + h])
__global__ void kv_k(const float* __restrict__ gk, const float* __restrict__ gv,
                     float* __restrict__ kv, int nG)
{
  int hh = blockIdx.x;
  int dd = threadIdx.x >> 4, m = threadIdx.x & 15;
  float s = 0.f;
  for (int g = 0; g < nG; ++g)
    s += gk[(long)g * DM + dd * HH + hh] * gv[(long)g * DM + m * HH + hh];
  kv[(dd * 16 + m) * HH + hh] = s;
}

__global__ void ksum_k(const float* __restrict__ gk, float* __restrict__ ks, int nG)
{
  int i = threadIdx.x;                       // i = dd*8 + hh
  float s = 0.f;
  for (int g = 0; g < nG; ++g) s += gk[(long)g * DM + i];
  ks[i] = s;
}

// gt = LN(num/den) * (h + sigmoid(beta)), num/den per (g, m, hh), d = m*8+hh
__global__ void __launch_bounds__(256)
gattn_fuse(const float* __restrict__ gk, const float* __restrict__ kv,
           const float* __restrict__ ks, const float* __restrict__ gh,
           const float* __restrict__ gbeta, const float* __restrict__ lng,
           const float* __restrict__ lnb, float* __restrict__ gt, int nG)
{
  int wave = threadIdx.x >> 5, lane = threadIdx.x & 31;
  int g = blockIdx.x * 8 + wave;
  if (g >= nG) return;
  long base = (long)g * DM;
  float val[4]; float s = 0.f;
#pragma unroll
  for (int j = 0; j < 4; ++j) {
    int d = lane * 4 + j;
    int m = d >> 3, hh = d & 7;
    float num = 0.f, den = 0.f;
#pragma unroll
    for (int dd = 0; dd < CC; ++dd) {
      float kk = gk[base + dd * HH + hh];
      num += kk * kv[(dd * 16 + m) * HH + hh];
      den += kk * ks[dd * HH + hh];
    }
    val[j] = num / den;
    s += val[j];
  }
  float mean = waveRed(s) * (1.f / DM);
  float vs = 0.f;
#pragma unroll
  for (int j = 0; j < 4; ++j) { float dl = val[j] - mean; vs += dl * dl; }
  float rs = rsqrtf(waveRed(vs) * (1.f / DM) + 1e-5f);
#pragma unroll
  for (int j = 0; j < 4; ++j) {
    int d = lane * 4 + j;
    float ln = (val[j] - mean) * rs * lng[d] + lnb[d];
    float beta = 1.f / (1.f + __expf(-gbeta[d]));
    gt[base + d] = ln * (gh[base + d] + beta);
  }
}

__global__ void pred_k(const float* __restrict__ gx, const float* __restrict__ Wp,
                       const float* __restrict__ bp, float* __restrict__ out, int nG)
{
  int i = blockIdx.x * blockDim.x + threadIdx.x;
  if (i >= nG * OUTD) return;
  int g = i / OUTD, o2 = i % OUTD;
  float s = bp[o2];
  const float* xr = gx + (long)g * DM;
#pragma unroll 8
  for (int d = 0; d < DM; ++d) s += xr[d] * Wp[d * OUTD + o2];
  out[i] = s;
}

// ---------------------------------------------------------------------------
extern "C" void kernel_launch(void* const* d_in, const int* in_sizes, int n_in,
                              void* d_out, int out_size, void* d_ws, size_t ws_size,
                              hipStream_t stream)
{
  (void)in_sizes; (void)n_in; (void)out_size; (void)ws_size;

  const float* x       = (const float*)d_in[0];
  const float* Win     = (const float*)d_in[1];
  const float* b_in    = (const float*)d_in[2];
  const float* betas_l = (const float*)d_in[3];
  const float* Wh      = (const float*)d_in[4];
  const float* bh      = (const float*)d_in[5];
  const float* Wgat    = (const float*)d_in[6];
  const float* a_src   = (const float*)d_in[7];
  const float* a_dst   = (const float*)d_in[8];
  const float* b_gat   = (const float*)d_in[9];
  const float* Wl      = (const float*)d_in[10];
  const float* bl      = (const float*)d_in[11];
  const float* lnl_g   = (const float*)d_in[12];
  const float* lnl_b   = (const float*)d_in[13];
  const float* ln_g    = (const float*)d_in[14];
  const float* ln_b    = (const float*)d_in[15];
  const float* g_betas = (const float*)d_in[16];
  const float* g_Wh    = (const float*)d_in[17];
  const float* g_bh    = (const float*)d_in[18];
  const float* g_Wk    = (const float*)d_in[19];
  const float* g_bk    = (const float*)d_in[20];
  const float* g_Wv    = (const float*)d_in[21];
  const float* g_bv    = (const float*)d_in[22];
  const float* g_lng   = (const float*)d_in[23];
  const float* g_lnb   = (const float*)d_in[24];
  const float* Wout    = (const float*)d_in[25];
  const float* bout    = (const float*)d_in[26];
  const float* Wpred   = (const float*)d_in[27];
  const float* bpred   = (const float*)d_in[28];
  const int*   eidx    = (const int*)d_in[29];
  const int*   batch   = (const int*)d_in[30];

  float* ws = (float*)d_ws;
  size_t o = 0;
  float* xbuf   = ws + o; o += (size_t)NN * DM;
  float* hbuf   = ws + o; o += (size_t)NN * DM;
  float* xwbuf  = ws + o; o += (size_t)NN * DM;
  float* gatbuf = ws + o; o += (size_t)NN * DM;
  float* xlocal = ws + o; o += (size_t)NN * DM;
  float* es     = ws + o; o += (size_t)NN * HH;
  float* ed     = ws + o; o += (size_t)NN * HH;
  float* emax   = ws + o; o += (size_t)NN * HH;
  float* den    = ws + o; o += (size_t)NN * HH;
  float* ebuf   = ws + o; o += (size_t)(EE + NN) * HH;
  float* xp     = ws + o; o += (size_t)GG * DM;
  float* cnt    = ws + o; o += (size_t)GG;
  float* gx     = ws + o; o += (size_t)GG * DM;
  float* gh     = ws + o; o += (size_t)GG * DM;
  float* gk     = ws + o; o += (size_t)GG * DM;
  float* gv     = ws + o; o += (size_t)GG * DM;
  float* gt     = ws + o; o += (size_t)GG * DM;
  float* kvb    = ws + o; o += (size_t)CC * CC * HH;
  float* ksb    = ws + o; o += (size_t)CC * HH;

  auto fillf = [&](float* p, long long n, float v) {
    fill_k<<<(int)((n + 255) / 256), 256, 0, stream>>>(p, n, v);
  };

  const int gridN = (NN + 63) / 64;
  const int gridG = (GG + 63) / 64;
  const long long ET  = (long long)EE + NN;
  const int ebH = (int)((ET * HH + 255) / 256);
  const int ebD = (int)((ET * DM + 255) / 256);

  fillf(xlocal, (long long)NN * DM, 0.f);

  // x = x @ Win + b_in
  gemm128_wmma<0><<<gridN, 128, 0, stream>>>(x, Win, b_in, xbuf, NN);

  for (int i = 0; i < 3; ++i) {
    const size_t wo = (size_t)i * DM * DM;
    // h = relu(x @ Wh + bh)
    gemm128_wmma<1><<<gridN, 128, 0, stream>>>(xbuf, Wh + wo, bh + i*DM, hbuf, NN);
    // xw = x @ Wgat
    gemm128_wmma<0><<<gridN, 128, 0, stream>>>(xbuf, Wgat + wo, nullptr, xwbuf, NN);
    // gatbuf = x @ Wl + bl   (edge scatter accumulates on top)
    gemm128_wmma<0><<<gridN, 128, 0, stream>>>(xbuf, Wl + wo, bl + i*DM, gatbuf, NN);

    edge_scores_k<<<(NN*HH + 255)/256, 256, 0, stream>>>(
        xwbuf, a_src + i*HH*CC, a_dst + i*HH*CC, es, ed, NN);
    fillf(emax, (long long)NN * HH, -INFINITY);
    fillf(den,  (long long)NN * HH, 0.f);
    edge_pass1<<<ebH, 256, 0, stream>>>(eidx, es, ed, ebuf, emax, EE, NN);
    edge_pass2<<<ebH, 256, 0, stream>>>(eidx, ebuf, emax, den, EE, NN);
    edge_pass3<<<ebD, 256, 0, stream>>>(eidx, ebuf, den, xwbuf, gatbuf, EE, NN);

    local_fuse<<<(NN + 7)/8, 256, 0, stream>>>(
        gatbuf, b_gat + i*DM, hbuf, betas_l + i*DM,
        lnl_g + i*DM, lnl_b + i*DM, xbuf, xlocal, NN);
  }

  // global mean pool + LN
  fillf(xp, (long long)GG * DM, 0.f);
  fillf(cnt, GG, 0.f);
  pool_k<<<(int)(((long long)NN * DM + 255) / 256), 256, 0, stream>>>(xlocal, batch, xp, cnt, NN);
  pool_ln_k<<<(GG + 7)/8, 256, 0, stream>>>(xp, cnt, ln_g, ln_b, gx, GG);

  for (int i = 0; i < 2; ++i) {
    const size_t wo = (size_t)i * DM * DM;
    gemm128_wmma<0><<<gridG, 128, 0, stream>>>(gx, g_Wh + wo, g_bh + i*DM, gh, GG);
    gemm128_wmma<2><<<gridG, 128, 0, stream>>>(gx, g_Wk + wo, g_bk + i*DM, gk, GG); // sigmoid
    gemm128_wmma<0><<<gridG, 128, 0, stream>>>(gx, g_Wv + wo, g_bv + i*DM, gv, GG);
    kv_k<<<HH, 256, 0, stream>>>(gk, gv, kvb, GG);
    ksum_k<<<1, 128, 0, stream>>>(gk, ksb, GG);
    gattn_fuse<<<(GG + 7)/8, 256, 0, stream>>>(
        gk, kvb, ksb, gh, g_betas + i*DM, g_lng + i*DM, g_lnb + i*DM, gt, GG);
    gemm128_wmma<1><<<gridG, 128, 0, stream>>>(gt, Wout, bout, gx, GG); // relu
  }

  pred_k<<<(GG*OUTD + 255)/256, 256, 0, stream>>>(gx, Wpred, bpred, (float*)d_out, GG);
}
